// CentralSpecificModel_60816736911416
// MI455X (gfx1250) — matrix operand order repaired and checked by reading, not compiled, not run on hardware
//
#include <hip/hip_runtime.h>
#include <cstdint>

// Problem constants (fixed by the reference).
#define NTOK   65536
#define NSPEC  4
#define DIN    256
#define DHID   1024
#define DOUT   256
#define TM     64          // rows per tile (one species per tile)
#define HC     64          // hidden-dim chunk
#define MAXT   (NTOK/TM + NSPEC)   // worst-case tile count (per-species padding)

// LDS layout (dword offsets inside one static allocation; base offset 0)
#define XOFF   0
#define W1OFF  (TM * (DIN / 2))            // 8192  dw
#define W2OFF  (W1OFF + HC * (DIN / 2))    // 16384 dw
#define HOFF   (W2OFF + DOUT * (HC / 2))   // 24576 dw
#define LDSTOT (HOFF + TM * (HC / 2))      // 26624 dw = 104 KB

typedef __attribute__((ext_vector_type(16))) __bf16 v16bf;
typedef __attribute__((ext_vector_type(8)))  float  v8f;

union FragBF { uint32_t u[8]; v16bf v; };

#if defined(__gfx1250__) && __has_builtin(__builtin_amdgcn_tensor_load_to_lds) && \
    __has_builtin(__builtin_amdgcn_s_wait_tensorcnt)
#define USE_TDM 1
#else
#define USE_TDM 0
#endif

__device__ __forceinline__ uint32_t pkbf(float a, float b) {
  uint32_t ua = __float_as_uint(a), ub = __float_as_uint(b);
  ua = (ua + 0x7fffu + ((ua >> 16) & 1u)) >> 16;          // RNE f32 -> bf16
  ub = (ub + 0x7fffu + ((ub >> 16) & 1u)) >> 16;
  return (ua & 0xffffu) | (ub << 16);
}
__device__ __forceinline__ uint16_t bfb(float a) {
  uint32_t u = __float_as_uint(a);
  return (uint16_t)((u + 0x7fffu + ((u >> 16) & 1u)) >> 16);
}

// ---- A-operand fragment (16x32 bf16) from row-major bf16 tile in LDS ----
// dword j of row m = elements K=2j,2j+1. Lanes 0-15: M=lane, K in {0..7,16..23};
// lanes 16-31: M=lane-16, K in {8..15,24..31}.  (ISA 7.12.2)
__device__ __forceinline__ v16bf load_a_frag(const uint32_t* lds, int row0,
                                             int strideDw, int jb, int half, int ml) {
  FragBF f;
  const uint32_t* p = lds + (size_t)(row0 + ml) * strideDw + jb + half * 4;
#pragma unroll
  for (int q = 0; q < 4; ++q) { f.u[q] = p[q]; f.u[q + 4] = p[q + 8]; }
  return f.v;
}

// ---- B-operand fragment (32x16 bf16) from [n][Kpair-dword] layout in LDS ----
__device__ __forceinline__ v16bf load_b_frag(const uint32_t* lds, int col0,
                                             int strideDw, int jb, int half, int nl) {
  FragBF f;
  const uint32_t* p = lds + (size_t)(col0 + nl) * strideDw + jb + half * 4;
#pragma unroll
  for (int q = 0; q < 4; ++q) { f.u[q] = p[q]; f.u[q + 4] = p[q + 8]; }
  return f.v;
}

#if USE_TDM
typedef uint32_t u32x4 __attribute__((ext_vector_type(4)));
typedef int      i32x8 __attribute__((ext_vector_type(8)));
typedef int      i32x4 __attribute__((ext_vector_type(4)));

// TDM 2D tile load: rows x tileDw dwords, tensor row stride rowStrideDw,
// packed contiguously into LDS at byte offset ldsByte. (ISA ch.8 D# layout)
__device__ __forceinline__ void tdm_load_2d(uint32_t ldsByte, const void* gsrc,
                                            uint32_t tileDw, uint32_t rows,
                                            uint32_t rowStrideDw, uint32_t tensorDw0) {
  uint64_t ga = (uint64_t)(uintptr_t)gsrc;
  u32x4 g0;
  g0[0] = 1u;                                   // count=1, user mode, no gather
  g0[1] = ldsByte;                              // lds_addr (bytes)
  g0[2] = (uint32_t)ga;                         // global_addr[31:0]
  g0[3] = (uint32_t)(ga >> 32) | (2u << 30);    // global_addr[56:32] | type=2
  i32x8 g1;
  g1[0] = (int)(2u << 16);                      // data_size = 4 bytes
  g1[1] = (int)((tensorDw0 & 0xffffu) << 16);                               // tensor_dim0 lo
  g1[2] = (int)(((tensorDw0 >> 16) & 0xffffu) | ((rows & 0xffffu) << 16));  // t0 hi | t1 lo
  g1[3] = (int)(((rows >> 16) & 0xffffu) | ((tileDw & 0xffffu) << 16));     // t1 hi | tile_dim0
  g1[4] = (int)(rows & 0xffffu);                // tile_dim1 | tile_dim2=0
  g1[5] = (int)rowStrideDw;                     // tensor_dim0_stride[31:0]
  g1[6] = 0;                                    // stride0 hi | stride1 lo
  g1[7] = 0;
  i32x4 z4 = {0, 0, 0, 0};                      // groups 2/3: dims 3/4 unused (2D)
  i32x8 z8 = {0, 0, 0, 0, 0, 0, 0, 0};
  __builtin_amdgcn_tensor_load_to_lds(g0, g1, z4, z4, z8, 0);
}
#endif

// ======================= routing (counting sort by species) =======================
__global__ void k_zero(int* ws) { if (threadIdx.x < 16) ws[threadIdx.x] = 0; }

__global__ void k_hist(const int* __restrict__ sp, int* __restrict__ cnt) {
  int i = blockIdx.x * blockDim.x + threadIdx.x;
  if (i < NTOK) atomicAdd(&cnt[sp[i] & 3], 1);
}

__global__ void k_scan(const int* __restrict__ cnt, int* __restrict__ cur,
                       int* __restrict__ ntiles, int* __restrict__ tsp,
                       int* __restrict__ tbs, int* __restrict__ tvc) {
  if (threadIdx.x != 0 || blockIdx.x != 0) return;
  int o = 0, nt = 0;
  for (int s = 0; s < NSPEC; ++s) {
    int c = cnt[s];
    cur[s] = o;                       // scatter cursor = segment start
    for (int t = 0; t * TM < c; ++t) {
      tsp[nt] = s;
      tbs[nt] = o + t * TM;
      int rem = c - t * TM;
      tvc[nt] = rem < TM ? rem : TM;
      ++nt;
    }
    o += c;
  }
  *ntiles = nt;
}

__global__ void k_scatter(const int* __restrict__ sp, int* __restrict__ cur,
                          int* __restrict__ ridx) {
  int i = blockIdx.x * blockDim.x + threadIdx.x;
  if (i < NTOK) {
    int p = atomicAdd(&cur[sp[i] & 3], 1);
    ridx[p] = i;
  }
}

// ============== weight pre-pack: f32 -> bf16 pairs in B-operand layout ==============
__global__ void k_packw1(const float* __restrict__ W1, uint32_t* __restrict__ w1p) {
  int idx = blockIdx.x * blockDim.x + threadIdx.x;      // 4*1024*128 = 2^19
  int j = idx & 127, n = (idx >> 7) & 1023, s = idx >> 17;
  const float* p = W1 + ((size_t)s * DIN + 2 * j) * DHID + n;
  w1p[idx] = pkbf(p[0], p[DHID]);
}
__global__ void k_packw2(const float* __restrict__ W2, uint32_t* __restrict__ w2p) {
  int idx = blockIdx.x * blockDim.x + threadIdx.x;      // 4*256*512 = 2^19
  int j = idx & 511, n = (idx >> 9) & 255, s = idx >> 17;
  const float* p = W2 + ((size_t)s * DHID + 2 * j) * DOUT + n;
  w2p[idx] = pkbf(p[0], p[DOUT]);
}

// ========================= fused grouped MoE GEMM =========================
__global__ __launch_bounds__(256)
void k_moe(const float* __restrict__ x, const float* __restrict__ b1,
           const float* __restrict__ b2, const uint32_t* __restrict__ w1p,
           const uint32_t* __restrict__ w2p, const int* __restrict__ ridx,
           const int* __restrict__ tsp, const int* __restrict__ tbs,
           const int* __restrict__ tvc, const int* __restrict__ ntiles,
           float* __restrict__ out) {
  __shared__ uint32_t ldsAll[LDSTOT];        // 104 KB, single allocation at LDS offset 0
  uint32_t* ldsX  = ldsAll + XOFF;           // 64 x 128 dw (bf16 [m][k])
  uint32_t* ldsW1 = ldsAll + W1OFF;          // 64 x 128 dw ([n][Kpair])
  uint32_t* ldsW2 = ldsAll + W2OFF;          // 256 x 32 dw ([n][Kpair])
  uint32_t* ldsH  = ldsAll + HOFF;           // 64 x 32 dw  (bf16 [m][k])

  const int blk = blockIdx.x;
  if (blk >= *ntiles) return;
  const int s = tsp[blk], base = tbs[blk], vcnt = tvc[blk];

  const int tid  = threadIdx.x;
  const int w    = tid >> 5, lane = tid & 31;
  const int half = lane >> 4, nl = lane & 15;
  const int rm   = w & 3;          // row sub-tile (16 rows each)
  const int wc   = w >> 2;         // column half

  const uint32_t* gW1 = w1p + (size_t)s * DHID * (DIN / 2);   // [1024][128]
  const uint32_t* gW2 = w2p + (size_t)s * DOUT * (DHID / 2);  // [256][512]

  // ---- stage X tile (gathered rows, f32 -> packed bf16) ----
  {
    int i = tid >> 2, part = tid & 3;                 // 64 rows x 4 col-quarters
    int r = (i < vcnt) ? ridx[base + i] : ridx[base];
    const float4* src = (const float4*)(x + (size_t)r * DIN + part * 64);
    uint32_t* dst = ldsX + (size_t)i * 128 + part * 32;
#pragma unroll
    for (int q = 0; q < 16; ++q) {
      float4 f = src[q];
      dst[2 * q]     = pkbf(f.x, f.y);
      dst[2 * q + 1] = pkbf(f.z, f.w);
    }
  }

  // output row gather + validity (C/D layout: m = v + 8*half)
  int rowg[8];
#pragma unroll
  for (int v = 0; v < 8; ++v) {
    int i = rm * 16 + v + half * 8;
    rowg[v] = (i < vcnt) ? ridx[base + i] : -1;
  }

  v8f yacc[8];
#pragma unroll
  for (int t = 0; t < 8; ++t) yacc[t] = 0.0f;

  __syncthreads();

  for (int ch = 0; ch < DHID / HC; ++ch) {
    // ---- stage W1/W2 chunks into LDS ----
#if USE_TDM
    if (w == 0) {
      // W1 chunk: 8192 contiguous dwords (one 32KB row)
      tdm_load_2d(W1OFF * 4, gW1 + (size_t)ch * HC * (DIN / 2),
                  HC * (DIN / 2), 1, HC * (DIN / 2), HC * (DIN / 2));
      // W2 chunk: 256 rows x 32 dwords, tensor row stride 512 dwords
      tdm_load_2d(W2OFF * 4, gW2 + ch * (HC / 2),
                  HC / 2, DOUT, DHID / 2, DHID / 2);
      __builtin_amdgcn_s_wait_tensorcnt(0);
    }
#else
    {
      const uint4* src = (const uint4*)(gW1 + (size_t)ch * HC * (DIN / 2));
      uint4* dst = (uint4*)ldsW1;
#pragma unroll
      for (int q = 0; q < 8; ++q) dst[(size_t)tid * 8 + q] = src[(size_t)tid * 8 + q];
    }
    {
      const uint4* src = (const uint4*)(gW2 + (size_t)tid * (DHID / 2) + ch * (HC / 2));
      uint4* dst = (uint4*)(ldsW2 + (size_t)tid * (HC / 2));
#pragma unroll
      for (int q = 0; q < 8; ++q) dst[q] = src[q];
    }
#endif
    // hint next chunk's weights toward the caches
    if (lane == 0 && ch + 1 < DHID / HC) {
      __builtin_prefetch(gW1 + (size_t)(ch + 1) * HC * (DIN / 2) + (size_t)w * 1024, 0, 0);
      __builtin_prefetch(gW2 + (size_t)(w * 32) * (DHID / 2) + (ch + 1) * (HC / 2), 0, 0);
    }
    __syncthreads();

    // ---- layer 1: H[64x64] = silu(X @ W1_chunk + b1_chunk), bf16 WMMA ----
    v8f hacc[2];
    hacc[0] = 0.0f; hacc[1] = 0.0f;
#pragma unroll
    for (int ks = 0; ks < DIN / 32; ++ks) {
      int jb = 16 * ks;
      v16bf a = load_a_frag(ldsX, rm * 16, DIN / 2, jb, half, nl);
#pragma unroll
      for (int t = 0; t < 2; ++t) {
        v16bf bf = load_b_frag(ldsW1, wc * 32 + t * 16, DIN / 2, jb, half, nl);
        hacc[t] = __builtin_amdgcn_wmma_f32_16x16x32_bf16(
            false, a, false, bf, (short)0, hacc[t], false, false);
      }
    }
    // bias + SiLU, write H back as row-major bf16 (A-operand layout for layer 2)
    uint16_t* hp = (uint16_t*)ldsH;
#pragma unroll
    for (int t = 0; t < 2; ++t) {
      int hc_local = wc * 32 + t * 16 + nl;
      float bias = b1[(size_t)s * DHID + ch * HC + hc_local];
#pragma unroll
      for (int v = 0; v < 8; ++v) {
        int m = rm * 16 + v + half * 8;
        float hv = hacc[t][v] + bias;
        hv = hv / (1.0f + __expf(-hv));                 // silu
        hp[(size_t)m * HC + hc_local] = bfb(hv);
      }
    }
    __syncthreads();

    // ---- layer 2: Y += H_chunk @ W2_chunk ----
#pragma unroll
    for (int ks = 0; ks < HC / 32; ++ks) {
      int jb = 16 * ks;
      v16bf a = load_a_frag(ldsH, rm * 16, HC / 2, jb, half, nl);
#pragma unroll
      for (int t = 0; t < 8; ++t) {
        v16bf bf = load_b_frag(ldsW2, wc * 128 + t * 16, HC / 2, jb, half, nl);
        yacc[t] = __builtin_amdgcn_wmma_f32_16x16x32_bf16(
            false, a, false, bf, (short)0, yacc[t], false, false);
      }
    }
    __syncthreads();
  }

  // ---- epilogue: +b2, scatter rows back ----
#pragma unroll
  for (int t = 0; t < 8; ++t) {
    int c = wc * 128 + t * 16 + nl;
    float bias = b2[(size_t)s * DOUT + c];
#pragma unroll
    for (int v = 0; v < 8; ++v) {
      if (rowg[v] >= 0) out[(size_t)rowg[v] * DOUT + c] = yacc[t][v] + bias;
    }
  }
}

// ================================ launcher ================================
extern "C" void kernel_launch(void* const* d_in, const int* in_sizes, int n_in,
                              void* d_out, int out_size, void* d_ws, size_t ws_size,
                              hipStream_t stream) {
  (void)in_sizes; (void)n_in; (void)out_size; (void)ws_size;
  const float* x  = (const float*)d_in[0];
  const int*   sp = (const int*)d_in[1];
  const float* W1 = (const float*)d_in[2];
  const float* b1 = (const float*)d_in[3];
  const float* W2 = (const float*)d_in[4];
  const float* b2 = (const float*)d_in[5];
  float* out = (float*)d_out;

  uint32_t* ws = (uint32_t*)d_ws;
  // dword-offset workspace map
  int* cnt = (int*)ws;                    // [4]
  int* cur = (int*)(ws + 4);              // [4]
  int* ntl = (int*)(ws + 8);              // [1]
  int* tsp = (int*)(ws + 2048);           // [MAXT]
  int* tbs = (int*)(ws + 4096);           // [MAXT]
  int* tvc = (int*)(ws + 6144);           // [MAXT]
  int* ridx = (int*)(ws + 8192);          // [NTOK]
  uint32_t* w1p = ws + 73728;             // 4*1024*128 dw = 2 MB
  uint32_t* w2p = ws + 73728 + 524288;    // 4*256*512  dw = 2 MB

  k_zero<<<1, 64, 0, stream>>>((int*)ws);
  k_hist<<<NTOK / 256, 256, 0, stream>>>(sp, cnt);
  k_scan<<<1, 1, 0, stream>>>(cnt, cur, ntl, tsp, tbs, tvc);
  k_scatter<<<NTOK / 256, 256, 0, stream>>>(sp, cur, ridx);
  k_packw1<<<2048, 256, 0, stream>>>(W1, w1p);
  k_packw2<<<2048, 256, 0, stream>>>(W2, w2p);
  k_moe<<<MAXT, 256, 0, stream>>>(x, b1, b2, w1p, w2p, ridx, tsp, tbs, tvc, ntl, out);
}
